// GPUDetections_20169166422187
// MI455X (gfx1250) — compile-verified, matrix operand order checked by reading, not run on hardware
//
#include <hip/hip_runtime.h>
#include <math.h>

#define CM     80      // foreground classes
#define NN     1000    // anchors / K (PRE_NMS == N)
#define NP     1008    // padded to 63*16 for 16-wide tiles
#define NT     63      // 16-wide tiles per dimension
#define NWORDS 64      // u16 mask words per row (63 used, padded) -> 128 B row stride
#define POSTN  100

typedef __attribute__((ext_vector_type(16))) _Float16     v16h;
typedef __attribute__((ext_vector_type(8)))  float        v8f;
typedef __attribute__((ext_vector_type(4)))  unsigned int u32x4;
typedef __attribute__((ext_vector_type(4)))  int          i32x4;
typedef __attribute__((ext_vector_type(8)))  int          i32x8;

// ---------------------------------------------------------------------------
// K1: softmax over 81 classes + per-class box decode, transposed to [b][c][n]
// ---------------------------------------------------------------------------
__global__ void k_decode(const float* __restrict__ cls, const float* __restrict__ box,
                         const float* __restrict__ anc, const float* __restrict__ info,
                         float* __restrict__ scores_t, float* __restrict__ boxes_t) {
  int bn = blockIdx.x;            // 0..1999  = b*NN + n
  int b  = bn / NN, n = bn % NN;
  __shared__ float e[81];
  __shared__ float red[2];
  int t = threadIdx.x;            // 128 threads
  const float* l = cls + (size_t)bn * 81;
  if (t < 81) e[t] = l[t];
  __syncthreads();
  if (t == 0) {
    float m = e[0];
    for (int i = 1; i < 81; ++i) m = fmaxf(m, e[i]);
    red[0] = m;
  }
  __syncthreads();
  if (t < 81) e[t] = __expf(e[t] - red[0]);
  __syncthreads();
  if (t == 0) {
    float s = 0.f;
    for (int i = 0; i < 81; ++i) s += e[i];
    red[1] = 1.0f / s;
  }
  __syncthreads();
  if (t < CM) {
    int c = t + 1;
    const float* a = anc + (size_t)bn * 4;
    float a0 = a[0], a1 = a[1], a2 = a[2], a3 = a[3];
    float ha = a2 - a0 + 1.f, wa = a3 - a1 + 1.f;
    float cya = a0 + 0.5f * ha, cxa = a1 + 0.5f * wa;
    const float* eb = box + (size_t)bn * 324 + c * 4;
    float dy = eb[0] * 0.1f, dx = eb[1] * 0.1f;     // /10
    float dh = eb[2] * 0.2f, dw = eb[3] * 0.2f;     // /5
    float cy = dy * ha + cya, cx = dx * wa + cxa;
    float h = __expf(dh) * ha, w = __expf(dw) * wa;
    float ymin = cy - 0.5f * h, xmin = cx - 0.5f * w;
    float ymax = cy + 0.5f * h - 1.f, xmax = cx + 0.5f * w - 1.f;
    float hm = info[b * 5 + 0] - 1.f, wm = info[b * 5 + 1] - 1.f;
    ymin = fminf(fmaxf(ymin, 0.f), hm);  ymax = fminf(fmaxf(ymax, 0.f), hm);
    xmin = fminf(fmaxf(xmin, 0.f), wm);  xmax = fminf(fmaxf(xmax, 0.f), wm);
    size_t o = (size_t)(b * CM + t) * NN + n;
    scores_t[o] = e[c] * red[1];
    float* ob = boxes_t + o * 4;
    ob[0] = ymin; ob[1] = xmin; ob[2] = ymax; ob[3] = xmax;
  }
}

// ---------------------------------------------------------------------------
// K2: per-(b,c) bitonic sort of 1024 (score,idx) in LDS, gather sorted boxes.
//     Sorted scores leave LDS via async LDS->global stores (ASYNCcnt path).
// ---------------------------------------------------------------------------
__global__ void k_sort(const float* __restrict__ scores_t, const float* __restrict__ boxes_t,
                       float* __restrict__ sscore, float* __restrict__ sboxes) {
  int p = blockIdx.x;             // 0..159
  __shared__ float s[1024];
  __shared__ int   id[1024];
  int t = threadIdx.x;            // 512 threads
  const float* sp = scores_t + (size_t)p * NN;
  for (int i = t; i < 1024; i += 512) { s[i] = (i < NN) ? sp[i] : -1e30f; id[i] = i; }
  __syncthreads();
  for (int k = 2; k <= 1024; k <<= 1)
    for (int j = k >> 1; j > 0; j >>= 1) {
      int i  = 2 * j * (t / j) + (t % j);
      int ix = i + j;
      bool desc = ((i & k) == 0);
      float si = s[i], sx = s[ix];
      if ((si < sx) == desc) {
        s[i] = sx; s[ix] = si;
        int tmp = id[i]; id[i] = id[ix]; id[ix] = tmp;
      }
      __syncthreads();
    }
  for (int kk = t; kk < NP; kk += 512) {
    float b0 = 0.f, b1 = 0.f, b2 = 0.f, b3 = 0.f;
    if (kk < NN) {
      // async copy sorted score straight from LDS to global (no VGPR bounce)
      float*   g = sscore + (size_t)p * NP + kk;
      unsigned l = (unsigned)(size_t)&s[kk];     // low 32 bits = LDS byte offset
      asm volatile("global_store_async_from_lds_b32 %0, %1, off"
                   :: "v"(g), "v"(l) : "memory");
      const float* src = boxes_t + ((size_t)p * NN + id[kk]) * 4;
      b0 = src[0]; b1 = src[1]; b2 = src[2]; b3 = src[3];
    }
    float* dst = sboxes + ((size_t)p * NP + kk) * 4;
    dst[0] = b0; dst[1] = b1; dst[2] = b2; dst[3] = b3;   // zero pad rows >= NN
  }
  asm volatile("s_wait_asynccnt 0x0" ::: "memory");
}

// ---------------------------------------------------------------------------
// K3: triangular 16x16-tiled overlap bitmask.
//     Boxes staged into LDS by the Tensor Data Mover (one D#, 16 KB tile).
//     WMMA supplies D[i][j] = (area_i + area_j)/64 (f16 in, f32 accum);
//     pair test: 1.5*inter > 32*D  <=>  IoU > 0.5   (division-free).
//     Wave32 ballot packs one 16x16 tile into 16 u16 mask words.
// ---------------------------------------------------------------------------
__global__ void k_bitmask(const float* __restrict__ sboxes, unsigned short* __restrict__ mask) {
  int p = blockIdx.x;             // 0..159
  __shared__ float bx[NP * 4];    // 16128 B, contiguous -> one TDM tile
  __shared__ float ar[NP];
  int t = threadIdx.x;            // 256 threads = 8 waves
  const float* src = sboxes + (size_t)p * NP * 4;

#if __has_builtin(__builtin_amdgcn_tensor_load_to_lds)
  if (t < 32) {                   // wave 0 issues one TDM descriptor
    unsigned long long ga = (unsigned long long)(size_t)src;
    unsigned ldsa = (unsigned)(size_t)&bx[0];
    // D# group0: count=1 | lds_addr | global_addr[56:0] | type=2
    u32x4 g0 = { 1u, ldsa, (unsigned)ga,
                 (unsigned)((ga >> 32) & 0x01FFFFFFu) | 0x80000000u };
    // D# group1: data_size=4B; tensor_dim0=tile_dim0=4032; dims1=1; strides=4032
    i32x8 g1 = { 0x20000,                 // data_size=2 (4B) at [17:16]
                 (int)(4032u << 16),      // tensor_dim0[15:0] at [79:64... 63:48]
                 (int)(1u << 16),         // tensor_dim0[31:16]=0, tensor_dim1[15:0]=1
                 (int)(4032u << 16),      // tensor_dim1[31:16]=0, tile_dim0=4032
                 1,                       // tile_dim1=1, tile_dim2=0
                 4032,                    // tensor_dim0_stride[31:0]
                 (int)(4032u << 16),      // dim0_stride[47:32]=0, dim1_stride[15:0]
                 0 };
    i32x4 gz4 = { 0, 0, 0, 0 };
#if __clang_major__ >= 23
    i32x8 gz8 = { 0, 0, 0, 0, 0, 0, 0, 0 };
    __builtin_amdgcn_tensor_load_to_lds(g0, g1, gz4, gz4, gz8, 0);
#else
    __builtin_amdgcn_tensor_load_to_lds(g0, g1, gz4, gz4, 0);
#endif
    __builtin_amdgcn_s_wait_tensorcnt(0);
  }
#else
  for (int i = t; i < NP * 4; i += 256) bx[i] = src[i];
#endif
  __syncthreads();
  for (int i = t; i < NP; i += 256)
    ar[i] = (bx[i * 4 + 2] - bx[i * 4 + 0]) * (bx[i * 4 + 3] - bx[i * 4 + 1]);
  __syncthreads();

  unsigned short* mrow = mask + (size_t)p * NP * NWORDS;
  int wv = t >> 5, lane = t & 31;
  const int NTILES = NT * (NT + 1) / 2;   // 2016, divisible by 8 -> uniform trip count
  for (int tile = wv; tile < NTILES; tile += 8) {
    int rem = tile, rT = 0;
    while (rem >= (NT - rT)) { rem -= (NT - rT); ++rT; }
    int cT = rT + rem;
    int rBase = rT * 16, cBase = cT * 16;

    // A[i][0]=area_i/64, A[i][1]=1 ; B[0][j]=1, B[1][j]=area_j/64
    v16h A = {}; v16h B = {};
    if (lane < 16) {
      A[0] = (_Float16)(ar[rBase + lane] * (1.f / 64.f));
      A[1] = (_Float16)1.0f;
      B[0] = (_Float16)1.0f;
      B[1] = (_Float16)(ar[cBase + lane] * (1.f / 64.f));
    }
    v8f Cc = {};
    v8f D = __builtin_amdgcn_wmma_f32_16x16x32_f16(false, A, false, B,
                                                   (short)0, Cc, false, false);
    // D vgpr r: lanes 0-15 -> row rBase+r, lanes 16-31 -> row rBase+r+8; col = lane&15
    int cj = cBase + (lane & 15);
    float c0 = bx[cj * 4 + 0], c1 = bx[cj * 4 + 1];
    float c2 = bx[cj * 4 + 2], c3 = bx[cj * 4 + 3];
    int rhalf = (lane >> 4) << 3;
    for (int r = 0; r < 8; ++r) {
      int ri = rBase + r + rhalf;
      float r0 = bx[ri * 4 + 0], r1 = bx[ri * 4 + 1];
      float r2 = bx[ri * 4 + 2], r3 = bx[ri * 4 + 3];
      float iy  = fmaxf(0.f, fminf(r2, c2) - fmaxf(r0, c0));
      float ixw = fmaxf(0.f, fminf(r3, c3) - fmaxf(r1, c1));
      float inter = iy * ixw;
      bool pred = (inter * 1.5f) > (32.f * D[r]);
      if (rT == cT) pred = pred && (cj > ri);   // only j > i suppresses
      unsigned bal = (unsigned)__ballot(pred ? 1 : 0);
      if (lane == 0) {
        mrow[(size_t)(rBase + r)     * NWORDS + cT] = (unsigned short)(bal & 0xffffu);
        mrow[(size_t)(rBase + r + 8) * NWORDS + cT] = (unsigned short)(bal >> 16);
      }
    }
  }
}

// ---------------------------------------------------------------------------
// K4: sequential greedy scan over the bitmask (per (b,c), 64 threads).
//     Mask rows are double-buffered in LDS via async global->LDS loads,
//     issued one row ahead to hide global latency on the serial loop.
// ---------------------------------------------------------------------------
__global__ void k_nms_scan(const unsigned short* __restrict__ mask,
                           const float* __restrict__ sscore,
                           float* __restrict__ flat) {
  int p = blockIdx.x; int b = p / CM, c = p % CM;
  __shared__ unsigned short removed[NWORDS];
  __shared__ unsigned short stage[2][NWORDS];   // double-buffered 128 B rows
  int t = threadIdx.x;            // 64 threads
  if (t < NWORDS) removed[t] = 0;
  const unsigned short* mrow = mask + (size_t)p * NP * NWORDS;

  // prefetch row 0 (32 lanes x b32 = 128 B)
  if (t < 32) {
    const void* g = (const char*)mrow + (size_t)t * 4;
    unsigned    l = (unsigned)(size_t)&stage[0][2 * t];
    asm volatile("global_load_async_to_lds_b32 %0, %1, off" :: "v"(l), "v"(g) : "memory");
  }
  for (int i = 0; i < NN; ++i) {
    asm volatile("s_wait_asynccnt 0x0" ::: "memory");   // row i landed (wave0)
    __syncthreads();                                    // visible to all waves
    bool sup = (removed[i >> 4] >> (i & 15)) & 1;       // uniform read
    __syncthreads();
    if (t < 32 && i + 1 < NN) {                         // prefetch row i+1
      const void* g = (const char*)mrow + (size_t)(i + 1) * (NWORDS * 2) + (size_t)t * 4;
      unsigned    l = (unsigned)(size_t)&stage[(i + 1) & 1][2 * t];
      asm volatile("global_load_async_to_lds_b32 %0, %1, off" :: "v"(l), "v"(g) : "memory");
    }
    if (!sup) {
      int w0 = i >> 4;
      if (t >= w0 && t < NT) removed[t] |= stage[i & 1][t];
    }
    __syncthreads();                                    // ORs + stage reads done
  }
  float* fb = flat + (size_t)b * (CM * NN) + (size_t)c * NN;
  const float* ss = sscore + (size_t)p * NP;
  for (int k = t; k < NN; k += 64) {
    bool sup = (removed[k >> 4] >> (k & 15)) & 1;
    fb[k] = sup ? 0.f : ss[k];
  }
}

// ---------------------------------------------------------------------------
// K5: per-batch top-100 by 100 argmax passes over the L2-resident flat array
// ---------------------------------------------------------------------------
__global__ void k_topk(float* __restrict__ flat, const float* __restrict__ sboxes,
                       float* __restrict__ out) {
  int b = blockIdx.x;             // 0..1
  __shared__ float vs[1024];
  __shared__ int   vi[1024];
  int t = threadIdx.x;            // 1024 threads
  float* fb = flat + (size_t)b * (CM * NN);
  for (int sel = 0; sel < POSTN; ++sel) {
    float best = -2e30f; int bidx = 0x7fffffff;
    for (int j = t; j < CM * NN; j += 1024) {
      if (j + 8192 < CM * NN) __builtin_prefetch(&fb[j + 8192], 0, 0);
      float s = fb[j];
      if (s > best) { best = s; bidx = j; }   // first occurrence wins ties
    }
    vs[t] = best; vi[t] = bidx;
    __syncthreads();
    for (int s = 512; s > 0; s >>= 1) {
      if (t < s) {
        float so = vs[t + s]; int io = vi[t + s];
        if (so > vs[t] || (so == vs[t] && io < vi[t])) { vs[t] = so; vi[t] = io; }
      }
      __syncthreads();
    }
    if (t == 0) {
      int idx = vi[0]; float sc = vs[0];
      int c = idx / NN, k = idx % NN;
      const float* bb = sboxes + ((size_t)(b * CM + c) * NP + k) * 4;
      float* o = out + (size_t)(b * POSTN + sel) * 6;
      o[0] = bb[0]; o[1] = bb[1]; o[2] = bb[2]; o[3] = bb[3];
      o[4] = sc; o[5] = (float)(c + 1);
      fb[idx] = -1e30f;           // exclude from later passes
    }
    __syncthreads();              // also makes fb[idx] write visible in-block
  }
}

// ---------------------------------------------------------------------------
extern "C" void kernel_launch(void* const* d_in, const int* in_sizes, int n_in,
                              void* d_out, int out_size, void* d_ws, size_t ws_size,
                              hipStream_t stream) {
  (void)in_sizes; (void)n_in; (void)out_size; (void)ws_size;
  const float* cls  = (const float*)d_in[0];  // [2,1000,81]
  const float* box  = (const float*)d_in[1];  // [2,1000,324]
  const float* anc  = (const float*)d_in[2];  // [2,1000,4]
  const float* info = (const float*)d_in[3];  // [2,5]
  float* out = (float*)d_out;                 // [2,100,6]

  char* ws = (char*)d_ws;
  size_t off = 0;
  auto alloc = [&](size_t bytes) -> void* {
    void* pp = ws + off;
    off += (bytes + 255) & ~(size_t)255;
    return pp;
  };
  float*          scores_t = (float*)alloc((size_t)2 * CM * NN * 4);            // 640 KB
  float*          boxes_t  = (float*)alloc((size_t)2 * CM * NN * 16);           // 2.5 MB
  float*          sscore   = (float*)alloc((size_t)2 * CM * NP * 4);            // 645 KB
  float*          sboxes   = (float*)alloc((size_t)2 * CM * NP * 16);           // 2.6 MB
  unsigned short* mask     = (unsigned short*)alloc((size_t)2 * CM * NP * NWORDS * 2); // 20.6 MB
  float*          flat     = (float*)alloc((size_t)2 * CM * NN * 4);            // 640 KB

  k_decode  <<<2 * NN, 128, 0, stream>>>(cls, box, anc, info, scores_t, boxes_t);
  k_sort    <<<2 * CM, 512, 0, stream>>>(scores_t, boxes_t, sscore, sboxes);
  k_bitmask <<<2 * CM, 256, 0, stream>>>(sboxes, mask);
  k_nms_scan<<<2 * CM,  64, 0, stream>>>(mask, sscore, flat);
  k_topk    <<<2,     1024, 0, stream>>>(flat, sboxes, out);
}
